// MotionHierarchyNode_21388937134589
// MI455X (gfx1250) — matrix operand order repaired and checked by reading, not compiled, not run on hardware
//
#include <hip/hip_runtime.h>

typedef __attribute__((ext_vector_type(16))) __bf16 v16bf;
typedef __attribute__((ext_vector_type(8)))  float  v8f;

#define NB 4
#define NTT 32
#define NN 2048
#define ND 3
#define NDM 256
#define NK 16

__device__ __forceinline__ __bf16 f2bf(float f) {
  unsigned u = __builtin_bit_cast(unsigned, f);
  unsigned r = u + 0x7FFFu + ((u >> 16) & 1u);   // round-to-nearest-even
  unsigned short h = (unsigned short)(r >> 16);
  return __builtin_bit_cast(__bf16, h);
}

// ---------------------------------------------------------------- mean + deltas
__global__ void k_mean_deltas(const float* __restrict__ pos,
                              float* __restrict__ mean_pos,
                              float* __restrict__ deltas) {
  int tid = blockIdx.x * blockDim.x + threadIdx.x;
  if (tid >= NB * NN * ND) return;
  int d = tid % ND, n = (tid / ND) % NN, b = tid / (ND * NN);
  float prev = pos[(((size_t)b * NTT + 0) * NN + n) * ND + d];
  float sum = prev;
  for (int t = 1; t < NTT; ++t) {
    float cur = pos[(((size_t)b * NTT + t) * NN + n) * ND + d];
    sum += cur;
    deltas[(((size_t)b * (NTT - 1) + (t - 1)) * NN + n) * ND + d] = cur - prev;
    prev = cur;
  }
  mean_pos[((size_t)b * NN + n) * ND + d] = sum * (1.0f / NTT);
}

// ---------------------------------------------------------------- kNN (top-16, lower index wins ties)
__global__ void k_knn(const float* __restrict__ mp, int* __restrict__ nbr) {
  int u = blockIdx.x * blockDim.x + threadIdx.x;
  if (u >= NB * NN) return;
  int b = u / NN, i = u % NN;
  const float* base = mp + (size_t)b * NN * ND;
  float xi = base[i * ND + 0], yi = base[i * ND + 1], zi = base[i * ND + 2];
  float bd[NK]; int bj[NK];
#pragma unroll
  for (int k = 0; k < NK; ++k) { bd[k] = __builtin_inff(); bj[k] = 0; }
  for (int j = 0; j < NN; ++j) {
    float dx = xi - base[j * ND], dy = yi - base[j * ND + 1], dz = zi - base[j * ND + 2];
    float dd = dx * dx + dy * dy + dz * dz;
    if (j == i) dd = 0.0f;
    if (dd < bd[NK - 1]) {
      float cd = dd; int cj = j;
#pragma unroll
      for (int s = 0; s < NK; ++s) {
        if (cd < bd[s]) { float td = bd[s]; int tj = bj[s]; bd[s] = cd; bj[s] = cj; cd = td; cj = tj; }
      }
    }
  }
#pragma unroll
  for (int k = 0; k < NK; ++k) nbr[(size_t)u * NK + k] = bj[k];
}

// ---------------------------------------------------------------- node MLP layer 1 (K=3, scalar)
__global__ void k_node_l1(const float* __restrict__ mp, const float* __restrict__ w,
                          const float* __restrict__ bias, float* __restrict__ outp) {
  int tid = blockIdx.x * blockDim.x + threadIdx.x;
  if (tid >= NB * NN * NDM) return;
  int c = tid & (NDM - 1), u = tid >> 8;
  float v = bias[c];
#pragma unroll
  for (int d = 0; d < ND; ++d) v += mp[u * ND + d] * w[d * NDM + c];
  outp[tid] = fmaxf(v, 0.f);
}

// ---------------------------------------------------------------- ew1 -> transposed bf16 (B-frag friendly)
__global__ void k_convert_wt(const float* __restrict__ w, __bf16* __restrict__ wt) {
  int tid = blockIdx.x * blockDim.x + threadIdx.x;
  if (tid >= NDM * NDM) return;
  int k = tid & (NDM - 1), n = tid >> 8;
  wt[tid] = f2bf(w[k * NDM + n]);  // wt[n][k] = w[k][n]
}

// ---------------------------------------------------------------- generic WMMA GEMM: C[M x 256] = act(A[M x Kd] @ W + b)
__global__ void __launch_bounds__(128) k_gemm_wmma(
    const float* __restrict__ A, const float* __restrict__ W,
    const float* __restrict__ bias, float* __restrict__ C,
    int M, int Kd, int act) {
  int wid = threadIdx.x >> 5, lane = threadIdx.x & 31;
  int lm = lane & 15, hi = lane >> 4;
  int tile = blockIdx.x * 4 + wid;
  int tiles = (M >> 4) << 4;            // (M/16) * (256/16)
  if (tile >= tiles) return;            // wave-uniform
  int mt = tile >> 4, nt = tile & 15;
  const float* arow = A + (size_t)(mt * 16 + lm) * Kd;
  int ncol = nt * 16 + lm;
  v8f acc = {0.f, 0.f, 0.f, 0.f, 0.f, 0.f, 0.f, 0.f};
  for (int k0 = 0; k0 < Kd; k0 += 32) {
    v16bf a, bfr;
    int c0 = k0 + hi * 16;
#pragma unroll
    for (int e = 0; e < 8; ++e) a[e] = f2bf(arow[c0 + e]);
#pragma unroll
    for (int e = 0; e < 8; ++e) a[8 + e] = f2bf(arow[c0 + 8 + e]);
    int kb = k0 + hi * 16;
#pragma unroll
    for (int e = 0; e < 16; ++e) bfr[e] = f2bf(W[(size_t)(kb + e) * NDM + ncol]);
    acc = __builtin_amdgcn_wmma_f32_16x16x32_bf16(false, a, false, bfr, (short)0, acc, false, false);
  }
  float bv = bias ? bias[ncol] : 0.f;
#pragma unroll
  for (int v = 0; v < 8; ++v) {
    int m = mt * 16 + hi * 8 + v;
    float val = acc[v] + bv;
    if (act) val = fmaxf(val, 0.f);
    C[(size_t)m * NDM + ncol] = val;
  }
}

// ---------------------------------------------------------------- fused edge MLP: hidden0 -> LDS -> WMMA x ew1 -> dot ew2
__global__ void __launch_bounds__(256) k_edge_mlp(
    const float* __restrict__ nf1, const float* __restrict__ nf2,
    const float* __restrict__ mp, const int* __restrict__ nbr,
    const float* __restrict__ ew0d, const float* __restrict__ eb0,
    const __bf16* __restrict__ ew1t, const float* __restrict__ eb1,
    const float* __restrict__ ew2, const float* __restrict__ eb2,
    float* __restrict__ elog) {
  __shared__ __bf16 h0s[8 * NK * NDM];          // 64 KB: 8 waves x (16 edges x 256)
  int wid = threadIdx.x >> 5, lane = threadIdx.x & 31;
  int lm = lane & 15, hi = lane >> 4;
  int u = blockIdx.x * 8 + wid;                 // node id (b*N+i), grid = 1024 -> 8192 nodes
  int b = u >> 11;
  int i = u & (NN - 1);
  const float* mpb = mp + (size_t)b * NN * ND;
  float xi = mpb[i * ND], yi = mpb[i * ND + 1], zi = mpb[i * ND + 2];
  __bf16* h0 = h0s + wid * (NK * NDM);
  const float* f1 = nf1 + (size_t)u * NDM;
  // ---- layer 0 (algebraically factored): h0[k][c] = relu(nf1[i] + nf2[j_k] + dp.ew0d + eb0)
  for (int k = 0; k < NK; ++k) {
    int j = nbr[(size_t)u * NK + k];
    const float* f2 = nf2 + (size_t)(b * NN + j) * NDM;
    float dx = xi - mpb[j * ND], dy = yi - mpb[j * ND + 1], dz = zi - mpb[j * ND + 2];
    for (int c = lane; c < NDM; c += 32) {
      float v = f1[c] + f2[c] + dx * ew0d[c] + dy * ew0d[NDM + c] + dz * ew0d[2 * NDM + c] + eb0[c];
      h0[k * NDM + c] = f2bf(fmaxf(v, 0.f));
    }
  }
  __syncthreads();
  // ---- layer 1: 16x256 @ 256x256 via WMMA bf16, 16 N-tiles accumulated in registers
  v8f acc[16];
  v8f z = {0.f, 0.f, 0.f, 0.f, 0.f, 0.f, 0.f, 0.f};
#pragma unroll
  for (int nt = 0; nt < 16; ++nt) acc[nt] = z;
  const __bf16* hrow = h0 + lm * NDM;
  for (int k0 = 0; k0 < NDM; k0 += 32) {
    v16bf a;
    int c0 = k0 + hi * 16;
#pragma unroll
    for (int e = 0; e < 8; ++e) a[e] = hrow[c0 + e];
#pragma unroll
    for (int e = 0; e < 8; ++e) a[8 + e] = hrow[c0 + 8 + e];
    int kb = k0 + hi * 16;
#pragma unroll
    for (int nt = 0; nt < 16; ++nt) {
      v16bf bfr;
      const __bf16* wr = ew1t + (size_t)(nt * 16 + lm) * NDM + kb;
#pragma unroll
      for (int e = 0; e < 16; ++e) bfr[e] = wr[e];
      acc[nt] = __builtin_amdgcn_wmma_f32_16x16x32_bf16(false, a, false, bfr, (short)0, acc[nt], false, false);
    }
  }
  // ---- layer 2 fused: logits[m] = sum_n relu(h1[m][n]+eb1[n]) * ew2[n] + eb2
  float s[8];
#pragma unroll
  for (int v = 0; v < 8; ++v) s[v] = 0.f;
#pragma unroll
  for (int nt = 0; nt < 16; ++nt) {
    int n = nt * 16 + lm;
    float bb = eb1[n], w2 = ew2[n];
#pragma unroll
    for (int v = 0; v < 8; ++v) {
      float h = fmaxf(acc[nt][v] + bb, 0.f);
      s[v] += h * w2;
    }
  }
#pragma unroll
  for (int off = 1; off < 16; off <<= 1) {
#pragma unroll
    for (int v = 0; v < 8; ++v) s[v] += __shfl_xor(s[v], off, 32);
  }
  if (lm == 0) {
    float b2 = eb2[0];
#pragma unroll
    for (int v = 0; v < 8; ++v) elog[(size_t)u * NK + hi * 8 + v] = s[v] + b2;
  }
}

// ---------------------------------------------------------------- softmax over K + aggregate + build concat input
__global__ void __launch_bounds__(256) k_soft_agg(
    const float* __restrict__ elog, const float* __restrict__ nf,
    const int* __restrict__ nbr, float* __restrict__ X) {
  int wid = threadIdx.x >> 5, lane = threadIdx.x & 31;
  int u = blockIdx.x * 8 + wid;
  int b = u >> 11;
  float l[NK]; int j[NK];
#pragma unroll
  for (int k = 0; k < NK; ++k) { l[k] = elog[(size_t)u * NK + k]; j[k] = nbr[(size_t)u * NK + k]; }
  float mx = l[0];
#pragma unroll
  for (int k = 1; k < NK; ++k) mx = fmaxf(mx, l[k]);
  float e[NK]; float ssum = 0.f;
#pragma unroll
  for (int k = 0; k < NK; ++k) { e[k] = expf(l[k] - mx); ssum += e[k]; }
  float inv = 1.f / ssum;
  for (int c = lane; c < NDM; c += 32) {
    float self = nf[(size_t)u * NDM + c];
    float a = 0.f;
#pragma unroll
    for (int k = 0; k < NK; ++k) a += e[k] * nf[(size_t)(b * NN + j[k]) * NDM + c];
    a *= inv;
    X[(size_t)u * 512 + c] = self;
    X[(size_t)u * 512 + NDM + c] = a;
  }
}

// ---------------------------------------------------------------- dense parent logits / probs / H + sparse probs
__global__ void __launch_bounds__(256) k_parent(
    const float* __restrict__ elog, const int* __restrict__ nbr,
    float* __restrict__ plog, float* __restrict__ pprob,
    float* __restrict__ Hout, float* __restrict__ pk) {
  int wid = threadIdx.x >> 5, lane = threadIdx.x & 31;
  int u = blockIdx.x * 8 + wid;
  float l[NK]; int j[NK];
#pragma unroll
  for (int k = 0; k < NK; ++k) { l[k] = elog[(size_t)u * NK + k]; j[k] = nbr[(size_t)u * NK + k]; }
  float mx = l[0];
#pragma unroll
  for (int k = 1; k < NK; ++k) mx = fmaxf(mx, l[k]);
  float e[NK]; float ssum = 0.f;
#pragma unroll
  for (int k = 0; k < NK; ++k) { e[k] = expf(l[k] - mx); ssum += e[k]; }
  float inv = 1.f / ssum;
  float p[NK];
#pragma unroll
  for (int k = 0; k < NK; ++k) p[k] = e[k] * inv;
  if (lane == 0) {
#pragma unroll
    for (int k = 0; k < NK; ++k) pk[(size_t)u * NK + k] = p[k];
  }
  float* lr = plog + (size_t)u * NN;
  float* pr = pprob + (size_t)u * NN;
  float* hr = Hout + (size_t)u * NN;
  const float NEGINF = -__builtin_inff();
  for (int col = lane; col < NN; col += 32) {
    float lg = NEGINF, pv = 0.f;
#pragma unroll
    for (int k = 0; k < NK; ++k) {
      if (j[k] == col) { lg = l[k]; pv = p[k]; }
    }
    lr[col] = lg; pr[col] = pv; hr[col] = pv;
  }
}

// ---------------------------------------------------------------- sparse H @ x  (x,y: (B,31,N,3))
__global__ void k_spmv(const float* __restrict__ x, const float* __restrict__ pk,
                       const int* __restrict__ nbr, float* __restrict__ y) {
  const int TOT = NB * (NTT - 1) * NN * ND;
  int tid = blockIdx.x * blockDim.x + threadIdx.x;
  if (tid >= TOT) return;
  int d = tid % ND;
  int i = (tid / ND) % NN;
  int t = (tid / (ND * NN)) % (NTT - 1);
  int b = tid / (ND * NN * (NTT - 1));
  int u = b * NN + i;
  const float* xb = x + (size_t)(b * (NTT - 1) + t) * NN * ND;
  float acc = 0.f;
#pragma unroll
  for (int k = 0; k < NK; ++k) {
    int j = nbr[(size_t)u * NK + k];
    acc += pk[(size_t)u * NK + k] * xb[j * ND + d];
  }
  y[tid] = acc;
}

__global__ void k_resid_init(const float* __restrict__ deltas, const float* __restrict__ hv,
                             float* __restrict__ out_resid, float* __restrict__ out_hat,
                             float* __restrict__ v1) {
  const int TOT = NB * (NTT - 1) * NN * ND;
  int tid = blockIdx.x * blockDim.x + threadIdx.x;
  if (tid >= TOT) return;
  float r = deltas[tid] - hv[tid];
  out_resid[tid] = r;
  out_hat[tid] = r;
  v1[tid] = r;
}

__global__ void k_hat_acc(const float* __restrict__ v2, float* __restrict__ out_hat) {
  const int TOT = NB * (NTT - 1) * NN * ND;
  int tid = blockIdx.x * blockDim.x + threadIdx.x;
  if (tid >= TOT) return;
  out_hat[tid] += v2[tid];
}

// ================================================================ host
extern "C" void kernel_launch(void* const* d_in, const int* in_sizes, int n_in,
                              void* d_out, int out_size, void* d_ws, size_t ws_size,
                              hipStream_t stream) {
  (void)in_sizes; (void)n_in; (void)out_size; (void)ws_size;
  const float* pos = (const float*)d_in[0];
  const float* nw0 = (const float*)d_in[1];  const float* nb0 = (const float*)d_in[2];
  const float* nw1 = (const float*)d_in[3];  const float* nb1 = (const float*)d_in[4];
  const float* ew0 = (const float*)d_in[5];  const float* eb0 = (const float*)d_in[6];
  const float* ew1 = (const float*)d_in[7];  const float* eb1 = (const float*)d_in[8];
  const float* ew2 = (const float*)d_in[9];  const float* eb2 = (const float*)d_in[10];
  const float* uw[2][2] = {{(const float*)d_in[11], (const float*)d_in[13]},
                           {(const float*)d_in[15], (const float*)d_in[17]}};
  const float* ub[2][2] = {{(const float*)d_in[12], (const float*)d_in[14]},
                           {(const float*)d_in[16], (const float*)d_in[18]}};

  float* out = (float*)d_out;
  const size_t SZ_D = (size_t)NB * (NTT - 1) * NN * ND;    // 761856
  const size_t SZ_H = (size_t)NB * NN * NN;                // 16777216
  float* out_deltas = out;
  float* out_hat    = out + SZ_D;
  float* out_resid  = out + 2 * SZ_D;
  float* out_H      = out + 3 * SZ_D;
  float* out_plog   = out_H + SZ_H;
  float* out_pprob  = out_plog + SZ_H;

  char* wp = (char*)d_ws;
  auto alloc = [&](size_t bytes) -> char* {
    char* p = wp; wp += (bytes + 255) & ~(size_t)255; return p;
  };
  float* mean_pos = (float*)alloc((size_t)NB * NN * ND * 4);
  int*   nbr      = (int*)  alloc((size_t)NB * NN * NK * 4);
  float* nfA      = (float*)alloc((size_t)NB * NN * NDM * 4);
  float* nfB      = (float*)alloc((size_t)NB * NN * NDM * 4);
  float* htmp     = (float*)alloc((size_t)NB * NN * NDM * 4);
  float* nf1      = (float*)alloc((size_t)NB * NN * NDM * 4);
  float* nf2      = (float*)alloc((size_t)NB * NN * NDM * 4);
  float* Xcat     = (float*)alloc((size_t)NB * NN * 512 * 4);
  float* elog     = (float*)alloc((size_t)NB * NN * NK * 4);
  float* pkbuf    = (float*)alloc((size_t)NB * NN * NK * 4);
  float* v1       = (float*)alloc(SZ_D * 4);
  float* v2       = (float*)alloc(SZ_D * 4);
  __bf16* ew1t    = (__bf16*)alloc((size_t)NDM * NDM * 2);

  // 1. mean / deltas
  k_mean_deltas<<<(NB * NN * ND + 255) / 256, 256, 0, stream>>>(pos, mean_pos, out_deltas);
  // 2. kNN
  k_knn<<<(NB * NN + 255) / 256, 256, 0, stream>>>(mean_pos, nbr);
  // 3. node MLP
  k_node_l1<<<(NB * NN * NDM + 255) / 256, 256, 0, stream>>>(mean_pos, nw0, nb0, htmp);
  {
    int M = NB * NN, tiles = M;
    k_gemm_wmma<<<(tiles + 3) / 4, 128, 0, stream>>>(htmp, nw1, nb1, nfA, M, NDM, 0);
  }
  // 4. one-time ew1 transpose+convert for the edge-kernel B fragments
  k_convert_wt<<<(NDM * NDM + 255) / 256, 256, 0, stream>>>(ew1, ew1t);

  float* nfc = nfA;
  float* nfn = nfB;
  const int M = NB * NN;         // 8192 node rows
  const int gemmBlocks = (M + 3) / 4;
  for (int layer = 0; layer < 3; ++layer) {
    // factored first edge layer: nf1 = nf @ ew0[0:256], nf2 = nf @ ew0[256:512]
    k_gemm_wmma<<<gemmBlocks, 128, 0, stream>>>(nfc, ew0, nullptr, nf1, M, NDM, 0);
    k_gemm_wmma<<<gemmBlocks, 128, 0, stream>>>(nfc, ew0 + 256 * NDM, nullptr, nf2, M, NDM, 0);
    // fused edge MLP (hidden0 in LDS -> WMMA x ew1 -> dot ew2)
    k_edge_mlp<<<M / 8, 256, 0, stream>>>(nf1, nf2, mean_pos, nbr,
                                          ew0 + 512 * NDM, eb0, ew1t, eb1, ew2, eb2, elog);
    if (layer < 2) {
      k_soft_agg<<<M / 8, 256, 0, stream>>>(elog, nfc, nbr, Xcat);
      k_gemm_wmma<<<gemmBlocks, 128, 0, stream>>>(Xcat, uw[layer][0], ub[layer][0], htmp, M, 512, 1);
      k_gemm_wmma<<<gemmBlocks, 128, 0, stream>>>(htmp, uw[layer][1], ub[layer][1], nfn, M, NDM, 0);
      float* t = nfc; nfc = nfn; nfn = t;
    }
  }

  // 5. dense parent logits / probs / H (+ sparse probs for SpMV)
  k_parent<<<M / 8, 256, 0, stream>>>(elog, nbr, out_plog, out_pprob, out_H, pkbuf);

  // 6. residual + truncated Neumann series (sparse H)
  const int TOT = (int)SZ_D;
  const int sb = (TOT + 255) / 256;
  k_spmv<<<sb, 256, 0, stream>>>(out_deltas, pkbuf, nbr, v1);        // H @ deltas
  k_resid_init<<<sb, 256, 0, stream>>>(out_deltas, v1, out_resid, out_hat, v1);
  float* va = v1; float* vb = v2;
  for (int level = 1; level <= 4; ++level) {
    k_spmv<<<sb, 256, 0, stream>>>(va, pkbuf, nbr, vb);              // v = H v
    k_hat_acc<<<sb, 256, 0, stream>>>(vb, out_hat);                  // hat += v
    float* t = va; va = vb; vb = t;
  }
}
